// LSTMModel_39161511805207
// MI455X (gfx1250) — matrix operand order repaired and checked by reading, not compiled, not run on hardware
//
#include <hip/hip_runtime.h>

// ---------------------------------------------------------------------------
// 2-layer LSTM (PyTorch semantics) + last-step FC for MI455X (gfx1250).
// f16 WMMA (16x16x32) with fp32 accumulate for all gate GEMMs.
// Layer-1 weight fragments + Whh2 fragments persist in VGPRs across the
// 200-step scan; Wih2 fragments stay in LDS (caps VGPR pressure < 256).
// h-state ping-pongs through LDS; c-state lives in registers in the WMMA
// C/D layout. Branch-free activations via v_exp_f32 / v_rcp_f32.
// ---------------------------------------------------------------------------

typedef _Float16 half_t;
typedef __attribute__((ext_vector_type(16))) _Float16 v16h;
typedef __attribute__((ext_vector_type(8)))  _Float16 v8h;
typedef __attribute__((ext_vector_type(8)))  float    v8f;

#define B_TOT 4096
#define T_LEN 200
#define F_IN  16
#define H1    55
#define H2    30
#define H1P   64      // padded hidden 1
#define H2P   32      // padded hidden 2
#define G1P   256     // 4*H1P gate columns, layer 1
#define G2P   128     // 4*H2P gate columns, layer 2

// dynamic LDS carve (byte offsets); all fragment bases 16B aligned
#define OFF_WB1I   0                          // [256][32] f16  (x -> gates1)
#define OFF_WBH1   (OFF_WB1I + G1P*32*2)      // [256][64] f16  (h1 -> gates1)
#define OFF_WB2I   (OFF_WBH1 + G1P*64*2)      // [128][64] f16  (h1 -> gates2)
#define OFF_WBH2   (OFF_WB2I + G2P*64*2)      // [128][32] f16  (h2 -> gates2)
#define OFF_H1BUF  (OFF_WBH2 + G2P*32*2)      // [2][16][64] f16 ping-pong
#define OFF_H2BUF  (OFF_H1BUF + 2*16*H1P*2)   // [2][16][32] f16 ping-pong
#define OFF_BIAS1  (OFF_H2BUF + 2*16*H2P*2)   // [256] f32 (bih+bhh, padded)
#define OFF_BIAS2  (OFF_BIAS1 + G1P*4)        // [128] f32
#define SMEM_BYTES (OFF_BIAS2 + G2P*4)        // = 81408 bytes

__device__ __forceinline__ v8f wmma16(v16h a, v16h b, v8f c) {
  // D = A(16x32 f16) * B(32x16 f16) + C(16x16 f32)
  return __builtin_amdgcn_wmma_f32_16x16x32_f16(
      /*neg_a=*/false, a, /*neg_b=*/false, b,
      /*c_mod=*/(short)0, c, /*reuse_a=*/false, /*reuse_b=*/false);
}

__device__ __forceinline__ v16h joinh(v8h lo, v8h hi) {
  union { v16h v; v8h h[2]; } u;
  u.h[0] = lo; u.h[1] = hi;
  return u.v;
}

// A fragment 16x32 f16 from an LDS tile stored row-major [16][kStride].
// ISA layout: lane l -> row m = l%16; half-group g = l/16;
// elements 0..7 = K kBase+g*8.., elements 8..15 = K kBase+16+g*8..
__device__ __forceinline__ v16h ldsA(const half_t* base, int lane, int kBase, int kStride) {
  const int m = lane & 15, g = lane >> 4;
  const half_t* p = base + m * kStride + kBase + g * 8;
  return joinh(*(const v8h*)p, *(const v8h*)(p + 16));
}

// B fragment 32x16 f16 from LDS weights stored [N][kStride] (i.e. W[n][k]).
// ISA layout: lane l -> col n = nBase + l%16; g = l/16;
// elements 0..7 = K kBase+g*8.., elements 8..15 = K kBase+16+g*8..
__device__ __forceinline__ v16h ldsB(const half_t* WB, int lane, int nBase, int kBase, int kStride) {
  const half_t* p = WB + (size_t)(nBase + (lane & 15)) * kStride + kBase + (lane >> 4) * 8;
  return joinh(*(const v8h*)p, *(const v8h*)(p + 16));
}

// A fragment for x_t: 16 real K columns (F=16), upper 16 zero-padded.
__device__ __forceinline__ v16h gmemA_x(const float* __restrict__ x, int b0, int t, int lane) {
  const int m = lane & 15, g = lane >> 4;
  const float* p = x + ((size_t)(b0 + m)) * T_LEN * F_IN + (size_t)t * F_IN + g * 8;
  v16h r = (v16h)(half_t)0.f;
#pragma unroll
  for (int i = 0; i < 8; ++i) r[i] = (half_t)p[i];   // K = g*8 .. g*8+7 (real)
  return r;                                          // elements 8..15 -> K 16..31 = 0
}

// Branch-free activations on the v_exp_f32 / v_rcp_f32 TRANS pipes.
#define LOG2E 1.4426950408889634f
__device__ __forceinline__ float fsig(float z) {
  return __builtin_amdgcn_rcpf(1.f + __builtin_amdgcn_exp2f(-LOG2E * z));
}
__device__ __forceinline__ float ftanh(float z) {
  return 1.f - 2.f * __builtin_amdgcn_rcpf(1.f + __builtin_amdgcn_exp2f((2.f * LOG2E) * z));
}

extern "C" __global__ __launch_bounds__(128)
void lstm2_wmma_kernel(const float* __restrict__ x,
                       const float* __restrict__ Wih1, const float* __restrict__ Whh1,
                       const float* __restrict__ bih1, const float* __restrict__ bhh1,
                       const float* __restrict__ Wih2, const float* __restrict__ Whh2,
                       const float* __restrict__ bih2, const float* __restrict__ bhh2,
                       const float* __restrict__ Wfc,  const float* __restrict__ bfc,
                       float* __restrict__ out)
{
  extern __shared__ char smem[];
  half_t* WB1i  = (half_t*)(smem + OFF_WB1I);
  half_t* WBh1  = (half_t*)(smem + OFF_WBH1);
  half_t* WB2i  = (half_t*)(smem + OFF_WB2I);
  half_t* WBh2  = (half_t*)(smem + OFF_WBH2);
  half_t* h1buf = (half_t*)(smem + OFF_H1BUF);
  half_t* h2buf = (half_t*)(smem + OFF_H2BUF);
  float*  bias1 = (float*)(smem + OFF_BIAS1);
  float*  bias2 = (float*)(smem + OFF_BIAS2);

  const int tid  = threadIdx.x;
  const int lane = tid & 31;
  const int wave = tid >> 5;
  const int b0   = blockIdx.x * 16;       // batch tile base (grid = 256)
  const int NT   = 128;                   // threads per block

  // ---- stage weights: f32 global -> zero-padded f16 [N][K] in LDS ----------
  for (int idx = tid; idx < G1P * 32; idx += NT) {
    int n = idx >> 5, k = idx & 31, g = n >> 6, r = n & 63;
    WB1i[idx] = (half_t)((r < H1 && k < F_IN) ? Wih1[(g * H1 + r) * F_IN + k] : 0.f);
  }
  for (int idx = tid; idx < G1P * 64; idx += NT) {
    int n = idx >> 6, k = idx & 63, g = n >> 6, r = n & 63;
    WBh1[idx] = (half_t)((r < H1 && k < H1) ? Whh1[(g * H1 + r) * H1 + k] : 0.f);
  }
  for (int idx = tid; idx < G2P * 64; idx += NT) {
    int n = idx >> 6, k = idx & 63, g = n >> 5, r = n & 31;
    WB2i[idx] = (half_t)((r < H2 && k < H1) ? Wih2[(g * H2 + r) * H1 + k] : 0.f);
  }
  for (int idx = tid; idx < G2P * 32; idx += NT) {
    int n = idx >> 5, k = idx & 31, g = n >> 5, r = n & 31;
    WBh2[idx] = (half_t)((r < H2 && k < H2) ? Whh2[(g * H2 + r) * H2 + k] : 0.f);
  }
  for (int n = tid; n < G1P; n += NT) {
    int g = n >> 6, r = n & 63;
    bias1[n] = (r < H1) ? bih1[g * H1 + r] + bhh1[g * H1 + r] : 0.f;
  }
  for (int n = tid; n < G2P; n += NT) {
    int g = n >> 5, r = n & 31;
    bias2[n] = (r < H2) ? bih2[g * H2 + r] + bhh2[g * H2 + r] : 0.f;
  }
  for (int i = tid; i < 16 * H1P; i += NT) h1buf[i] = (half_t)0.f;   // parity 0 = h0
  for (int i = tid; i < 16 * H2P; i += NT) h2buf[i] = (half_t)0.f;
  __syncthreads();

  // wave w owns hidden cols [16w,16w+16) of every layer-1 gate;
  // waves 0,1 own hidden cols [16w,16w+16) of every layer-2 gate.
  const int hc1 = wave * 16;
  const int hc2 = (wave & 1) * 16;
  const int mrow = (lane >> 4) * 8;       // C/D layout: rows mrow..mrow+7 this lane

  float b1g[4], b2g[4];
#pragma unroll
  for (int g = 0; g < 4; ++g) {
    b1g[g] = bias1[g * H1P + hc1 + (lane & 15)];
    b2g[g] = bias2[g * H2P + hc2 + (lane & 15)];
  }

  // ---- hoist loop-invariant weight B-fragments into VGPRs -----------------
  v16h B1x[4], B1h0[4], B1h1[4];          // 12 frags = 96 VGPRs (all waves)
#pragma unroll
  for (int g = 0; g < 4; ++g) {
    const int nB = g * H1P + hc1;
    B1x[g]  = ldsB(WB1i, lane, nB, 0,  32);
    B1h0[g] = ldsB(WBh1, lane, nB, 0,  64);
    B1h1[g] = ldsB(WBh1, lane, nB, 32, 64);
  }
  v16h B2h[4] = {};                       // Whh2 frags, 32 VGPRs (waves 0,1)
  if (wave < 2) {
#pragma unroll
    for (int g = 0; g < 4; ++g)
      B2h[g] = ldsB(WBh2, lane, g * H2P + hc2, 0, 32);
  }

  v8f c1 = (v8f)0.f;                      // cell state, C/D layout (this wave's 16 cols)
  v8f c2 = (v8f)0.f;                      // layer-2 cell state (waves 0,1)

  for (int t = 0; t < T_LEN; ++t) {
    const int rp = t & 1, wp = rp ^ 1;
    const half_t* h1r = h1buf + rp * 16 * H1P;
    half_t*       h1w = h1buf + wp * 16 * H1P;
    const half_t* h2r = h2buf + rp * 16 * H2P;
    half_t*       h2w = h2buf + wp * 16 * H2P;

    // ---------------- layer 1: gates = x_t*Wih1^T + h1*Whh1^T + b ----------
    v16h ax = gmemA_x(x, b0, t, lane);
    if (t + 1 < T_LEN) {                  // prefetch next timestep of x
      const float* pn = x + ((size_t)(b0 + (lane & 15))) * T_LEN * F_IN
                          + (size_t)(t + 1) * F_IN + (lane >> 4) * 8;
      __builtin_prefetch(pn, 0, 1);
    }
    v16h ah0 = ldsA(h1r, lane, 0,  H1P);
    v16h ah1 = ldsA(h1r, lane, 32, H1P);

    v8f acc[4];
#pragma unroll
    for (int g = 0; g < 4; ++g) {
      v8f c = (v8f)b1g[g];                // bias pre-loaded into accumulator
      c = wmma16(ax,  B1x[g],  c);
      c = wmma16(ah0, B1h0[g], c);
      c = wmma16(ah1, B1h1[g], c);
      acc[g] = c;
    }
#pragma unroll
    for (int e = 0; e < 8; ++e) {         // i,f,g,o -> c,h  (C/D layout)
      float iv = fsig (acc[0][e]);
      float fv = fsig (acc[1][e]);
      float gv = ftanh(acc[2][e]);
      float ov = fsig (acc[3][e]);
      float cv = fv * c1[e] + iv * gv;
      c1[e] = cv;
      float hv = ov * ftanh(cv);
      h1w[(mrow + e) * H1P + hc1 + (lane & 15)] = (half_t)hv;
    }
    __syncthreads();                      // h1(t) complete & old buffers free

    // ---------------- layer 2 (waves 0,1): gates = h1*Wih2^T + h2*Whh2^T ---
    if (wave < 2) {
      v16h a0 = ldsA(h1w, lane, 0,  H1P);
      v16h a1 = ldsA(h1w, lane, 32, H1P);
      v16h ah = ldsA(h2r, lane, 0,  H2P);
      v8f acc2[4];
#pragma unroll
      for (int g = 0; g < 4; ++g) {
        const int nB = g * H2P + hc2;
        v8f c = (v8f)b2g[g];
        c = wmma16(a0, ldsB(WB2i, lane, nB, 0,  64), c);
        c = wmma16(a1, ldsB(WB2i, lane, nB, 32, 64), c);
        c = wmma16(ah, B2h[g], c);
        acc2[g] = c;
      }
#pragma unroll
      for (int e = 0; e < 8; ++e) {
        float iv = fsig (acc2[0][e]);
        float fv = fsig (acc2[1][e]);
        float gv = ftanh(acc2[2][e]);
        float ov = fsig (acc2[3][e]);
        float cv = fv * c2[e] + iv * gv;
        c2[e] = cv;
        float hv = ov * ftanh(cv);
        h2w[(mrow + e) * H2P + hc2 + (lane & 15)] = (half_t)hv;
      }
    }
    __syncthreads();                      // h2(t) complete before next step
  }

  // ---------------- final FC: out[16,6] = h2_last * Wfc^T + bfc ------------
  const half_t* hf = h2buf + (T_LEN & 1) * 16 * H2P;  // parity of last write
  for (int idx = tid; idx < 16 * 6; idx += NT) {
    int m = idx / 6, o = idx % 6;
    float s = bfc[o];
#pragma unroll
    for (int k = 0; k < H2; ++k) s += (float)hf[m * H2P + k] * Wfc[o * H2 + k];
    out[(size_t)(b0 + m) * 6 + o] = s;
  }
}

extern "C" void kernel_launch(void* const* d_in, const int* in_sizes, int n_in,
                              void* d_out, int out_size, void* d_ws, size_t ws_size,
                              hipStream_t stream) {
  (void)in_sizes; (void)n_in; (void)d_ws; (void)ws_size; (void)out_size;
  const float* x    = (const float*)d_in[0];
  const float* Wih1 = (const float*)d_in[1];
  const float* Whh1 = (const float*)d_in[2];
  const float* bih1 = (const float*)d_in[3];
  const float* bhh1 = (const float*)d_in[4];
  const float* Wih2 = (const float*)d_in[5];
  const float* Whh2 = (const float*)d_in[6];
  const float* bih2 = (const float*)d_in[7];
  const float* bhh2 = (const float*)d_in[8];
  const float* Wfc  = (const float*)d_in[9];
  const float* bfc  = (const float*)d_in[10];
  float* out = (float*)d_out;

  dim3 grid(B_TOT / 16);   // 256 batch tiles
  dim3 block(128);         // 4 wave32 per tile
  lstm2_wmma_kernel<<<grid, block, SMEM_BYTES, stream>>>(
      x, Wih1, Whh1, bih1, bhh1, Wih2, Whh2, bih2, bhh2, Wfc, bfc, out);
}
